// SoftClusteringBlock_82051055222995
// MI455X (gfx1250) — matrix operand order repaired and checked by reading, not compiled, not run on hardware
//
#include <hip/hip_runtime.h>

// ---------------------------------------------------------------------------
// Fused soft-clustering block for MI455X (gfx1250, wave32, WMMA).
//   cross  = X @ C^T          (bf16 WMMA, f32 accum)
//   dist   = sqrt(max(|x|^2 + |c|^2 - 2*cross, 0))
//   assoc  = softmax(-dist)   (online/flash-style, never materialized)
//   out    = assoc @ C        (bf16 WMMA, f32 accum)
// B=8, N=8192, F=128, K=512  ->  65536 rows.
// ---------------------------------------------------------------------------

typedef __attribute__((ext_vector_type(16))) __bf16 v16bf;
typedef __attribute__((ext_vector_type(8)))  float  v8f;

struct Frag32B { uint4 lo, hi; };   // 32 bytes == one v16bf fragment per lane

static constexpr int Fdim = 128;
static constexpr int Kdim = 512;
static constexpr int ROWS_PER_WG = 128;   // 8 waves * 16 rows
static constexpr int WAVES = 8;

// LDS layout (bytes)
static constexpr int B1_BYTES   = (Kdim / 16) * (Fdim / 32) * 32 * 32; // 131072
static constexpr int B2_BYTES   = (Kdim / 32) * (Fdim / 16) * 32 * 32; // 131072
static constexpr int CSQ_OFF    = B1_BYTES + B2_BYTES;                  // 262144
static constexpr int CSQ_BYTES  = Kdim * 4;                             // 2048
static constexpr int STAGE_OFF  = CSQ_OFF + CSQ_BYTES;                  // 264192
static constexpr int STAGE_BYTES = WAVES * 16 * 32 * 2;                 // 8192
static constexpr int SMEM_BYTES = STAGE_OFF + STAGE_BYTES;              // 272384

__device__ __forceinline__ v16bf ldsFrag(const char* p) {
    Frag32B f;
    f.lo = *(const uint4*)(p);
    f.hi = *(const uint4*)(p + 16);
    return __builtin_bit_cast(v16bf, f);
}

__global__ void __launch_bounds__(256, 1)
soft_cluster_fused(const float* __restrict__ x,
                   const float* __restrict__ c,
                   float* __restrict__ out)
{
    extern __shared__ char smem[];
    char*  b1    = smem;                       // GEMM1 B fragments (C^T, f x k)
    char*  b2    = smem + B1_BYTES;            // GEMM2 B fragments (C,   k x f)
    float* csq   = (float*)(smem + CSQ_OFF);   // |c_k|^2
    __bf16* stage = (__bf16*)(smem + STAGE_OFF);

    const int tid   = threadIdx.x;
    const int lane  = tid & 31;
    const int wave  = tid >> 5;
    const int lhalf = lane >> 4;   // 0 / 1
    const int lmod  = lane & 15;

    // ---- |c_k|^2 ----
    for (int k = tid; k < Kdim; k += 256) {
        const float4* row = (const float4*)(c + (size_t)k * Fdim);
        float s = 0.f;
        #pragma unroll 4
        for (int i = 0; i < Fdim / 4; ++i) {
            float4 v = row[i];
            s += v.x * v.x + v.y * v.y + v.z * v.z + v.w * v.w;
        }
        csq[k] = s;
    }

    // ---- b1: B fragments of C^T (32 f-rows x 16 k-cols per fragment).
    // B k order used here (contiguous, same as A): lanes 0-15 hold K(=f)-rows 0..15, lanes 16-31 hold 16..31;
    // element e of the v16bf -> K = 16*lhalf + e, N(=k) = lmod. Contiguous in f.
    for (int s = wave; s < 128; s += WAVES) {
        const int kj = s >> 2, fi = s & 3;
        const int k  = kj * 16 + lmod;
        const int f0 = fi * 32 + lhalf * 16;
        const float4* src = (const float4*)(c + (size_t)k * Fdim + f0);
        v16bf v;
        #pragma unroll
        for (int i = 0; i < 4; ++i) {
            float4 t = src[i];
            v[i * 4 + 0] = (__bf16)t.x; v[i * 4 + 1] = (__bf16)t.y;
            v[i * 4 + 2] = (__bf16)t.z; v[i * 4 + 3] = (__bf16)t.w;
        }
        Frag32B fr = __builtin_bit_cast(Frag32B, v);
        uint4* dst = (uint4*)(b1 + ((size_t)s * 32 + lane) * 32);
        dst[0] = fr.lo; dst[1] = fr.hi;
    }

    // ---- b2: B fragments of C (32 k-rows x 16 f-cols): K = 16*lhalf + e, N(=f) = lmod.
    for (int s = wave; s < 128; s += WAVES) {
        const int kc = s >> 3, fj = s & 7;
        const int f  = fj * 16 + lmod;
        const int k0 = kc * 32 + lhalf * 16;
        v16bf v;
        #pragma unroll
        for (int e = 0; e < 16; ++e)
            v[e] = (__bf16)c[(size_t)(k0 + e) * Fdim + f];
        Frag32B fr = __builtin_bit_cast(Frag32B, v);
        uint4* dst = (uint4*)(b2 + ((size_t)s * 32 + lane) * 32);
        dst[0] = fr.lo; dst[1] = fr.hi;
    }

    __syncthreads();

    // ---- per-wave 16-row tile ----
    const int rowBase = blockIdx.x * ROWS_PER_WG + wave * 16;
    const int m = rowBase + lmod;

    // A fragments of x (16x32 bf16 each): lane=row lmod; element e covers
// f = fi*32 + lhalf*16 + e  -> contiguous 16 floats.
    v16bf aFrag[4];
    float xsqPart = 0.f;
    #pragma unroll
    for (int fi = 0; fi < 4; ++fi) {
        const int f0 = fi * 32 + lhalf * 16;
        const float4* p0 = (const float4*)(x + (size_t)m * Fdim + f0);
        const float4* p1 = (const float4*)(x + (size_t)m * Fdim + f0 + 8);
        float4 r0 = p0[0], r1 = p0[1], r2 = p1[0], r3 = p1[1];
        float vals[16] = { r0.x, r0.y, r0.z, r0.w, r1.x, r1.y, r1.z, r1.w,
                           r2.x, r2.y, r2.z, r2.w, r3.x, r3.y, r3.z, r3.w };
        v16bf v;
        #pragma unroll
        for (int e = 0; e < 16; ++e) {
            v[e] = (__bf16)vals[e];
            xsqPart += vals[e] * vals[e];
        }
        aFrag[fi] = v;
    }
    // lanes l and l^16 together cover all 128 features of row lmod
    float xsqRow = xsqPart + __shfl_xor(xsqPart, 16, 32);
    float xsqj[8];
    #pragma unroll
    for (int j = 0; j < 8; ++j)                 // C/D row of slot j is j + 8*lhalf
        xsqj[j] = __shfl(xsqRow, j + 8 * lhalf, 32);

    v8f oFrag[8] = {};
    float mRun[8], sRun[8];
    #pragma unroll
    for (int j = 0; j < 8; ++j) { mRun[j] = -3.0e38f; sRun[j] = 0.f; }

    __bf16* myStage = stage + wave * (16 * 32);

    for (int kc = 0; kc < Kdim / 32; ++kc) {
        float zA[8], zB[8];
        #pragma unroll
        for (int sub = 0; sub < 2; ++sub) {
            const int k16 = kc * 2 + sub;                 // 16-wide k chunk
            v8f cross = {};
            #pragma unroll
            for (int fi = 0; fi < 4; ++fi) {
                v16bf bfr = ldsFrag(b1 + ((size_t)(k16 * 4 + fi) * 32 + lane) * 32);
                cross = __builtin_amdgcn_wmma_f32_16x16x32_bf16(
                    false, aFrag[fi], false, bfr, (short)0, cross, false, false);
            }
            const float csql = csq[k16 * 16 + lmod];      // lane = k column
            float* z = sub ? zB : zA;
            #pragma unroll
            for (int j = 0; j < 8; ++j) {
                float d2 = xsqj[j] + csql - 2.f * cross[j];
                z[j] = -sqrtf(fmaxf(d2, 0.f));
            }
        }

        // online softmax over this 32-wide chunk (columns live across 16 lanes)
        #pragma unroll
        for (int j = 0; j < 8; ++j) {
            float zm = fmaxf(zA[j], zB[j]);
            #pragma unroll
            for (int msk = 1; msk < 16; msk <<= 1)
                zm = fmaxf(zm, __shfl_xor(zm, msk, 32));
            const float mNew  = fmaxf(mRun[j], zm);
            const float scale = __expf(mRun[j] - mNew);
            const float pA = __expf(zA[j] - mNew);
            const float pB = __expf(zB[j] - mNew);
            float ps = pA + pB;
            #pragma unroll
            for (int msk = 1; msk < 16; msk <<= 1)
                ps += __shfl_xor(ps, msk, 32);
            sRun[j] = sRun[j] * scale + ps;
            mRun[j] = mNew;
            #pragma unroll
            for (int fj = 0; fj < 8; ++fj) oFrag[fj][j] *= scale;
            const int row = j + 8 * lhalf;                // D layout -> stage (row major)
            myStage[row * 32 + lmod]      = (__bf16)pA;
            myStage[row * 32 + 16 + lmod] = (__bf16)pB;
        }

        // transpose-staged probabilities back as a 16x32 A fragment
        {
            Frag32B fr;
            fr.lo = *(const uint4*)(myStage + lmod * 32 + lhalf * 16);
            fr.hi = *(const uint4*)(myStage + lmod * 32 + lhalf * 16 + 8);
            v16bf aP = __builtin_bit_cast(v16bf, fr);
            #pragma unroll
            for (int fj = 0; fj < 8; ++fj) {
                v16bf bfr = ldsFrag(b2 + ((size_t)(kc * 8 + fj) * 32 + lane) * 32);
                oFrag[fj] = __builtin_amdgcn_wmma_f32_16x16x32_bf16(
                    false, aP, false, bfr, (short)0, oFrag[fj], false, false);
            }
        }
    }

    // normalize by the softmax denominator and store
    #pragma unroll
    for (int j = 0; j < 8; ++j) {
        const float rinv = 1.f / sRun[j];
        const size_t row = (size_t)(rowBase + j + 8 * lhalf);
        #pragma unroll
        for (int fj = 0; fj < 8; ++fj)
            out[row * Fdim + fj * 16 + lmod] = oFrag[fj][j] * rinv;
    }
}

extern "C" void kernel_launch(void* const* d_in, const int* in_sizes, int n_in,
                              void* d_out, int out_size, void* d_ws, size_t ws_size,
                              hipStream_t stream) {
    const float* x = (const float*)d_in[0];          // [B, N, F] f32
    const float* c = (const float*)d_in[1];          // [K, F]    f32
    float* out = (float*)d_out;                      // [B, N, F] f32

    const int rows   = in_sizes[0] / Fdim;           // B*N = 65536
    const int blocks = rows / ROWS_PER_WG;           // 512

    (void)hipFuncSetAttribute((const void*)soft_cluster_fused,
                              hipFuncAttributeMaxDynamicSharedMemorySize,
                              SMEM_BYTES);
    soft_cluster_fused<<<blocks, 256, SMEM_BYTES, stream>>>(x, c, out);
}
